// MyGNN_14345190769012
// MI455X (gfx1250) — compile-verified, matrix operand order checked by reading, not run on hardware
//
#include <hip/hip_runtime.h>
#include <math.h>

// ---------------------------------------------------------------------------
// GraphSAGE 2-layer forward for MI455X (gfx1250, wave32, WMMA).
//
// Sizes (fixed by the reference):
#define NNODES 100000
#define NEDGES 1600000
#define FIN    128
#define FHID   256
#define FOUT   128
//
// Strategy:
//  * Irregular aggregation: fp32 global atomics into a dense [N,128] buffer.
//    Whole feature matrix (51MB) fits in the 192MB L2, so scatter traffic
//    (~3.2GB total for both layers) is L2/HBM bound and cheap at 23.3TB/s.
//  * All GEMMs via v_wmma_f32_16x16x32_f16 (f16 inputs, fp32 accumulate).
//    One wave computes one 16x16 output tile; 100000/16 = 6250 exact row
//    tiles, Nc multiples of 64 -> no partial tiles, EXEC stays all-ones.
//  * SAGE layers fuse both products (mean@Wl^T and x@Wr^T) into one
//    accumulator chain per tile.
// ---------------------------------------------------------------------------

typedef _Float16 v16h __attribute__((ext_vector_type(16)));
typedef float    v8f  __attribute__((ext_vector_type(8)));

// ------------------------------- utility kernels ---------------------------

__global__ void k_fill_zero(float* __restrict__ p, size_t n) {
  size_t i = (size_t)blockIdx.x * blockDim.x + threadIdx.x;
  size_t st = (size_t)gridDim.x * blockDim.x;
  for (; i < n; i += st) p[i] = 0.0f;
}

__global__ void k_cast_f16(_Float16* __restrict__ dst, const float* __restrict__ src,
                           size_t n) {
  size_t i = (size_t)blockIdx.x * blockDim.x + threadIdx.x;
  size_t st = (size_t)gridDim.x * blockDim.x;
  for (; i < n; i += st) dst[i] = (_Float16)src[i];
}

// degree: cnt[tgt[e]] += 1
__global__ void k_degree(float* __restrict__ cnt, const int* __restrict__ tgt, int E) {
  int e = blockIdx.x * blockDim.x + threadIdx.x;
  if (e < E) atomicAdd(&cnt[tgt[e]], 1.0f);
}

// scatter-add: agg[tgt[e], :] += x[src[e], :]   (feature dim fixed at 128)
// one thread handles 4 consecutive features (float4 gather, 4 atomics).
__global__ void k_scatter128(float* __restrict__ agg, const float* __restrict__ x,
                             const int* __restrict__ src, const int* __restrict__ tgt,
                             int E) {
  int tid = blockIdx.x * blockDim.x + threadIdx.x;
  if (tid >= E * 32) return;
  int e = tid >> 5;
  int c = (tid & 31) << 2;
  int s = src[e];
  int t = tgt[e];
  float4 v = *reinterpret_cast<const float4*>(x + (size_t)s * 128 + c);
  float* d = agg + (size_t)t * 128 + c;
  atomicAdd(d + 0, v.x);
  atomicAdd(d + 1, v.y);
  atomicAdd(d + 2, v.z);
  atomicAdd(d + 3, v.w);
}

// mean = agg / max(cnt,1), cast to f16.  n = N*128, feature dim 128 (>>7).
__global__ void k_mean_cast(_Float16* __restrict__ mB, const float* __restrict__ agg,
                            const float* __restrict__ cnt, size_t n) {
  size_t i = (size_t)blockIdx.x * blockDim.x + threadIdx.x;
  size_t st = (size_t)gridDim.x * blockDim.x;
  for (; i < n; i += st) {
    float c = fmaxf(cnt[i >> 7], 1.0f);
    mB[i] = (_Float16)(agg[i] / c);
  }
}

// --------------------------------- WMMA GEMM -------------------------------
// ISA 16-bit A-fragment layout (16x32 tile, row-major source, ld = K):
//   lanes 0-15 : row = lane,    halves K= k0+0..7  then k0+16..23
//   lanes16-31 : row = lane-16, halves K= k0+8..15 then k0+24..31
// B (K x 16) column n == row n of W[out,K] row-major -> identical packing.
__device__ __forceinline__ v16h load_frag(const _Float16* __restrict__ base,
                                          int row, int ld, int k0, int lane) {
  int kh = (lane & 16) ? 8 : 0;
  const _Float16* p = base + (size_t)row * ld + (k0 + kh);
  union { v16h v; uint4 q[2]; } u;
  u.q[0] = *reinterpret_cast<const uint4*>(p);       // K = k0+kh   .. +7
  u.q[1] = *reinterpret_cast<const uint4*>(p + 16);  // K = k0+kh+16.. +23
  return u.v;
}

// out[M,Nc] = A1[M,K] @ W1[Nc,K]^T + bias  (+ A2 @ W2^T)   (optional relu)
// grid: (M/16, Nc/64), block: 128 (4 waves, one 16-col tile each).
__global__ __launch_bounds__(128) void k_sage_gemm(
    float* __restrict__ out,
    const _Float16* __restrict__ A1, const _Float16* __restrict__ W1,
    const float* __restrict__ bias,
    const _Float16* __restrict__ A2, const _Float16* __restrict__ W2,
    int K, int Nc, int relu) {
  int lane = threadIdx.x & 31;
  int wave = threadIdx.x >> 5;
  int rowBase = blockIdx.x * 16;
  int colBase = (blockIdx.y * 4 + wave) * 16;
  int arow = rowBase + (lane & 15);
  int wrow = colBase + (lane & 15);

  float bv = bias[colBase + (lane & 15)];
  v8f acc;
#pragma unroll
  for (int i = 0; i < 8; ++i) acc[i] = bv;

  for (int k0 = 0; k0 < K; k0 += 32) {
    v16h a = load_frag(A1, arow, K, k0, lane);
    v16h b = load_frag(W1, wrow, K, k0, lane);
    acc = __builtin_amdgcn_wmma_f32_16x16x32_f16(false, a, false, b,
                                                 (short)0, acc, false, false);
  }
  if (A2 != nullptr) {
    for (int k0 = 0; k0 < K; k0 += 32) {
      v16h a = load_frag(A2, arow, K, k0, lane);
      v16h b = load_frag(W2, wrow, K, k0, lane);
      acc = __builtin_amdgcn_wmma_f32_16x16x32_f16(false, a, false, b,
                                                   (short)0, acc, false, false);
    }
  }

  // C/D layout: VGPR i -> M = i (lanes 0-15) or i+8 (lanes 16-31); N = lane&15
  int col = colBase + (lane & 15);
  int r0  = rowBase + ((lane & 16) ? 8 : 0);
#pragma unroll
  for (int i = 0; i < 8; ++i) {
    float v = acc[i];
    if (relu) v = fmaxf(v, 0.0f);
    out[(size_t)(r0 + i) * Nc + col] = v;
  }
}

// ------------------------------ row L2 normalize ---------------------------
// one block per row, blockDim == C, writes f16 normalized row.
__global__ void k_l2norm_f16(_Float16* __restrict__ dst, const float* __restrict__ src,
                             int C) {
  extern __shared__ float s[];
  int r = blockIdx.x, t = threadIdx.x;
  float v = src[(size_t)r * C + t];
  s[t] = v * v;
  __syncthreads();
  for (int off = C >> 1; off > 0; off >>= 1) {
    if (t < off) s[t] += s[t + off];
    __syncthreads();
  }
  float scale = 1.0f / fmaxf(sqrtf(s[0]), 1e-12f);
  dst[(size_t)r * C + t] = (_Float16)(v * scale);
}

// --------------- final: normalize y, linear2 (128->2), softmax -------------
// d_out layout: probs [N,2] then y [N,128] (reference returns (probs, y)).
__global__ __launch_bounds__(128) void k_final(float* __restrict__ out,
                                               const float* __restrict__ ypre,
                                               const float* __restrict__ Wlin2,
                                               const float* __restrict__ blin2) {
  __shared__ float s[128];
  int r = blockIdx.x, t = threadIdx.x;
  float v = ypre[(size_t)r * 128 + t];
  s[t] = v * v;
  __syncthreads();
  for (int off = 64; off > 0; off >>= 1) {
    if (t < off) s[t] += s[t + off];
    __syncthreads();
  }
  float scale = 1.0f / fmaxf(sqrtf(s[0]), 1e-12f);
  float y = v * scale;
  out[(size_t)NNODES * 2 + (size_t)r * 128 + t] = y;
  __syncthreads();

  s[t] = y * Wlin2[t];
  __syncthreads();
  for (int off = 64; off > 0; off >>= 1) {
    if (t < off) s[t] += s[t + off];
    __syncthreads();
  }
  float l0 = s[0] + blin2[0];
  __syncthreads();

  s[t] = y * Wlin2[128 + t];
  __syncthreads();
  for (int off = 64; off > 0; off >>= 1) {
    if (t < off) s[t] += s[t + off];
    __syncthreads();
  }
  float l1 = s[0] + blin2[1];

  if (t == 0) {
    float m = fmaxf(l0, l1);
    float e0 = expf(l0 - m), e1 = expf(l1 - m);
    float inv = 1.0f / (e0 + e1);
    out[(size_t)r * 2 + 0] = e0 * inv;
    out[(size_t)r * 2 + 1] = e1 * inv;
  }
}

// ---------------------------------------------------------------------------

extern "C" void kernel_launch(void* const* d_in, const int* in_sizes, int n_in,
                              void* d_out, int out_size, void* d_ws, size_t ws_size,
                              hipStream_t stream) {
  (void)in_sizes; (void)n_in; (void)out_size; (void)ws_size;

  const float* x     = (const float*)d_in[0];
  const int*   ei    = (const int*)d_in[1];      // [2,E]: src row then tgt row
  const float* W1l   = (const float*)d_in[2];
  const float* b1l   = (const float*)d_in[3];
  const float* W1r   = (const float*)d_in[4];
  const float* Wlin  = (const float*)d_in[5];
  const float* blin  = (const float*)d_in[6];
  const float* W2l   = (const float*)d_in[7];
  const float* b2l   = (const float*)d_in[8];
  const float* W2r   = (const float*)d_in[9];
  const float* Wlin2 = (const float*)d_in[10];
  const float* blin2 = (const float*)d_in[11];
  const int* src = ei;
  const int* tgt = ei + NEDGES;
  float* out = (float*)d_out;

  // workspace layout (fp32 region then f16 region) -- ~385 MB total
  float* cnt  = (float*)d_ws;                       // N
  float* agg  = cnt + NNODES;                       // N*128 (reused both layers)
  float* h1   = agg + (size_t)NNODES * 128;         // N*256
  float* h2   = h1  + (size_t)NNODES * 256;         // N*128
  float* ypre = h2  + (size_t)NNODES * 128;         // N*128
  _Float16* xB   = (_Float16*)(ypre + (size_t)NNODES * 128);  // N*128
  _Float16* mB   = xB  + (size_t)NNODES * 128;                // N*128 (reused)
  _Float16* hNB  = mB  + (size_t)NNODES * 128;                // N*256
  _Float16* h2B  = hNB + (size_t)NNODES * 256;                // N*128
  _Float16* W1lB = h2B + (size_t)NNODES * 128;                // 256*128
  _Float16* W1rB = W1lB + 256 * 128;
  _Float16* WlinB= W1rB + 256 * 128;                          // 128*256
  _Float16* W2lB = WlinB + 128 * 256;                         // 128*128
  _Float16* W2rB = W2lB + 128 * 128;                          // 128*128

  const int T = 256;
  dim3 gsCast(4096), bT(T);

  // 0) zero cnt+agg (contiguous)
  k_fill_zero<<<2048, T, 0, stream>>>(cnt, (size_t)NNODES * 129);

  // 1) casts to f16
  k_cast_f16<<<gsCast, bT, 0, stream>>>(xB,    x,    (size_t)NNODES * 128);
  k_cast_f16<<<64,     bT, 0, stream>>>(W1lB,  W1l,  256 * 128);
  k_cast_f16<<<64,     bT, 0, stream>>>(W1rB,  W1r,  256 * 128);
  k_cast_f16<<<64,     bT, 0, stream>>>(WlinB, Wlin, 128 * 256);
  k_cast_f16<<<64,     bT, 0, stream>>>(W2lB,  W2l,  128 * 128);
  k_cast_f16<<<64,     bT, 0, stream>>>(W2rB,  W2r,  128 * 128);

  // 2) degrees
  k_degree<<<(NEDGES + T - 1) / T, bT, 0, stream>>>(cnt, tgt, NEDGES);

  // 3) layer 1 aggregation: agg += x[src] at tgt ; mean -> f16
  k_scatter128<<<(NEDGES * 32 + T - 1) / T, bT, 0, stream>>>(agg, x, src, tgt, NEDGES);
  k_mean_cast<<<gsCast, bT, 0, stream>>>(mB, agg, cnt, (size_t)NNODES * 128);

  // 4) conv1: h1 = mean@W1l^T + b1l + x@W1r^T   [N,256]
  k_sage_gemm<<<dim3(NNODES / 16, FHID / 64), 128, 0, stream>>>(
      h1, mB, W1lB, b1l, xB, W1rB, FIN, FHID, 0);

  // 5) L2-normalize rows(256) -> f16
  k_l2norm_f16<<<NNODES, FHID, FHID * sizeof(float), stream>>>(hNB, h1, FHID);

  // 6) linear + relu: h2 = relu(hN@Wlin^T + blin)   [N,128]
  k_sage_gemm<<<dim3(NNODES / 16, FOUT / 64), 128, 0, stream>>>(
      h2, hNB, WlinB, blin, nullptr, nullptr, FHID, FOUT, 1);
  k_cast_f16<<<gsCast, bT, 0, stream>>>(h2B, h2, (size_t)NNODES * 128);

  // 7) layer 2 aggregation (reuse agg)
  k_fill_zero<<<2048, T, 0, stream>>>(agg, (size_t)NNODES * 128);
  k_scatter128<<<(NEDGES * 32 + T - 1) / T, bT, 0, stream>>>(agg, h2, src, tgt, NEDGES);
  k_mean_cast<<<gsCast, bT, 0, stream>>>(mB, agg, cnt, (size_t)NNODES * 128);

  // 8) conv2: ypre = mean@W2l^T + b2l + h2@W2r^T   [N,128]
  k_sage_gemm<<<dim3(NNODES / 16, FOUT / 64), 128, 0, stream>>>(
      ypre, mB, W2lB, b2l, h2B, W2rB, FOUT, FOUT, 0);

  // 9) normalize -> y, linear2 -> softmax -> probs ; write d_out
  k_final<<<NNODES, 128, 0, stream>>>(out, ypre, Wlin2, blin2);
}
